// SMOLUpdate_4063039062769
// MI455X (gfx1250) — matrix operand order repaired and checked by reading, not compiled, not run on hardware
//
#include <hip/hip_runtime.h>
#include <hip/hip_bf16.h>
#include <math.h>

typedef __attribute__((ext_vector_type(16))) _Float16 v16h;
typedef __attribute__((ext_vector_type(8)))  _Float16 v8h;
typedef __attribute__((ext_vector_type(8)))  float    v8f;

#define NRES 4096
#define SS   128
#define NA0  5
#define NNP  11
#define AA   16      // NA0 + NNP
#define LL   256
#define PP   64
#define KA   320     // 64 (dirs padded from 48) + 256 (rbf)

// ---------------- LDS layout (bytes) ----------------
static constexpr unsigned OFF_A    = 0;                          // 128 x 320 f16 (dirs|pad|rbf), row-major
static constexpr unsigned OFF_B    = OFF_A   + 128u*KA*2u;       // 320 x 64  f16, frag-swizzled
static constexpr unsigned OFF_PF   = OFF_B   + KA*64u*2u;        // 128 x 64  f32 (pair, reused)
static constexpr unsigned OFF_PH   = OFF_PF  + 128u*64u*4u;      // 128 x 64  f16 (post-LN), row-major
static constexpr unsigned OFF_W1   = OFF_PH  + 128u*64u*2u;      // 64 x 128  f16, frag-swizzled
static constexpr unsigned OFF_ACT  = OFF_W1  + 64u*128u*2u;      // 128 x 128 f16, row-major
static constexpr unsigned OFF_W2   = OFF_ACT + 128u*128u*2u;     // 128 x 64  f16, frag-swizzled
static constexpr unsigned OFF_LOC  = OFF_W2  + 128u*64u*2u;      // 256 f32  local row
static constexpr unsigned OFF_SPOS = OFF_LOC + 256u*4u;          // 384 f32  smol_pos row
static constexpr unsigned OFF_PALL = OFF_SPOS+ 384u*4u;          // 48  f32  pos_all
static constexpr unsigned OFF_PL   = OFF_PALL+ 48u*4u;           // 36  f32  p_local
static constexpr unsigned OFF_FRM  = OFF_PL  + 36u*4u;           // 12  f32  e1,e2,e3,ca
static constexpr unsigned OFF_BSL  = OFF_FRM + 12u*4u;           // 64  f32  local@w_local
static constexpr unsigned OFF_GATE = OFF_BSL + 64u*4u;           // 64  f32  local@w_gate
static constexpr unsigned OFF_LNS  = OFF_GATE+ 64u*4u;           // 64
static constexpr unsigned OFF_LNO  = OFF_LNS + 64u*4u;           // 64
static constexpr unsigned OFF_B1   = OFF_LNO + 64u*4u;           // 128
static constexpr unsigned OFF_B2   = OFF_B1  + 128u*4u;          // 64
static constexpr unsigned OFF_WT   = OFF_B2  + 64u*4u;           // 7*64 f32 w_type
static constexpr unsigned OFF_MSK  = OFF_WT  + 7u*64u*4u;        // 128 f32 mask
static constexpr unsigned OFF_H    = OFF_MSK + 128u*4u;          // 64  f32 gated mean
static constexpr unsigned OFF_ST   = OFF_H   + 64u*4u;           // 128 i32 smol_type
static constexpr unsigned OFF_CNT  = OFF_ST  + 128u*4u;          // 1 f32
static constexpr unsigned SMEM_BYTES = OFF_CNT + 16u;            // ~245 KB <= 320 KB/WGP

__device__ __forceinline__ float gelu_f(float x) {
  float x3 = x * x * x;
  return 0.5f * x * (1.0f + tanhf(0.7978845608028654f * (x + 0.044715f * x3)));
}

__device__ __forceinline__ v16h cat8(v8h lo, v8h hi) {
  return __builtin_shufflevector(lo, hi, 0,1,2,3,4,5,6,7,8,9,10,11,12,13,14,15);
}

// A fragment (16x32 f16), row-major source. This code uses contiguous k order: the per-lane
// elements are contiguous: elems 0..15 = row[k..k+15]
// (caller offsets row by kb + koffA, koffA = 0 for lanes 0-15, 16 for 16-31).
// -> two ds_load_b128.
__device__ __forceinline__ v16h load_frag_a(const _Float16* __restrict row) {
  v8h lo = *(const v8h*)(row);
  v8h hi = *(const v8h*)(row + 8);
  return cat8(lo, hi);
}

// B fragment from frag-swizzled storage: chunk (kt, col, grp) holds 16
// contiguous halves = K rows kb+grp*16 .. +15 of column col. -> two ds_load_b128.
__device__ __forceinline__ v16h load_frag_b_swz(const _Float16* __restrict base,
                                                int kt, int ncol, int col, int grp) {
  const _Float16* q = base + (((kt * ncol + col) << 1) + grp) * 16;
  v8h lo = *(const v8h*)(q);
  v8h hi = *(const v8h*)(q + 8);
  return cat8(lo, hi);
}

// swizzled store index for element (k, col) of an ncol-wide B matrix
__device__ __forceinline__ int bswz(int k, int col, int ncol) {
  int kt = k >> 5, grp = (k >> 4) & 1, j = k & 15;
  return (((kt * ncol + col) << 1) + grp) * 16 + j;
}

__global__ __launch_bounds__(256)
void smol_fused_kernel(const float* __restrict localG,   // (N,256)
                       const float* __restrict posG,     // (N,5,3)
                       const int*   __restrict stypeG,   // (N,128)
                       const float* __restrict sposG,    // (N,128,3)
                       const unsigned char* __restrict smaskG, // (N,128) bool
                       const float* __restrict wPointsG, // (256,33)
                       const float* __restrict wTypeG,   // (7,64)
                       const float* __restrict wLocalG,  // (256,64)
                       const float* __restrict wDirG,    // (48,64)
                       const float* __restrict wDistG,   // (256,64)
                       const float* __restrict lnScaleG, // (64)
                       const float* __restrict lnOffG,   // (64)
                       const float* __restrict wMlp1G,   // (64,128)
                       const float* __restrict bMlp1G,   // (128)
                       const float* __restrict wMlp2G,   // (128,64)
                       const float* __restrict bMlp2G,   // (64)
                       const float* __restrict wGateG,   // (256,64)
                       const float* __restrict wOutG,    // (64,256)
                       float* __restrict outG)           // (N,256)
{
  extern __shared__ char smem[];
  _Float16* aS   = (_Float16*)(smem + OFF_A);
  _Float16* bS   = (_Float16*)(smem + OFF_B);
  float*    pairF= (float*)   (smem + OFF_PF);
  _Float16* pH   = (_Float16*)(smem + OFF_PH);
  _Float16* w1S  = (_Float16*)(smem + OFF_W1);
  _Float16* actS = (_Float16*)(smem + OFF_ACT);
  _Float16* w2S  = (_Float16*)(smem + OFF_W2);
  float*    locS = (float*)(smem + OFF_LOC);
  float*    sposS= (float*)(smem + OFF_SPOS);
  float*    pallS= (float*)(smem + OFF_PALL);
  float*    plS  = (float*)(smem + OFF_PL);
  float*    frm  = (float*)(smem + OFF_FRM);
  float*    bslS = (float*)(smem + OFF_BSL);
  float*    gateS= (float*)(smem + OFF_GATE);
  float*    lnS  = (float*)(smem + OFF_LNS);
  float*    lnO  = (float*)(smem + OFF_LNO);
  float*    b1S  = (float*)(smem + OFF_B1);
  float*    b2S  = (float*)(smem + OFF_B2);
  float*    wtS  = (float*)(smem + OFF_WT);
  float*    mskS = (float*)(smem + OFF_MSK);
  float*    hS   = (float*)(smem + OFF_H);
  int*      stS  = (int*)  (smem + OFF_ST);
  float*    cntS = (float*)(smem + OFF_CNT);

  const int n    = blockIdx.x;
  const int tid  = threadIdx.x;
  const int wave = tid >> 5;
  const int lane = tid & 31;
  const int lane15 = lane & 15;
  const int koffA = (lane < 16) ? 0 : 16;
  const int grp   = (lane < 16) ? 0 : 1;

  // ---- phase 1: stage small per-residue inputs + constants -------------------
  locS[tid] = localG[n*LL + tid];
  if (tid < 64) { lnS[tid] = lnScaleG[tid]; lnO[tid] = lnOffG[tid]; b2S[tid] = bMlp2G[tid]; }
  if (tid < 128) {
    b1S[tid]  = bMlp1G[tid];
    stS[tid]  = stypeG[n*SS + tid];
    mskS[tid] = smaskG[n*SS + tid] ? 1.0f : 0.0f;
  }
  for (int i = tid; i < SS*3; i += 256) sposS[i] = sposG[n*SS*3 + i];
  for (int i = tid; i < 7*PP; i += 256) wtS[i] = wTypeG[i];
  __syncthreads();

  // ---- phase 2: p_local, frames, local projections ---------------------------
  if (tid < 33) {                 // p_local = local @ w_points
    float s = 0.f;
    for (int i = 0; i < LL; ++i) s += locS[i] * wPointsG[i*33 + tid];
    plS[tid] = s;
  }
  if (tid >= 64 && tid < 128) {   // baseLocal = local @ w_local
    int p = tid - 64; float s = 0.f;
    for (int i = 0; i < LL; ++i) s += locS[i] * wLocalG[i*PP + p];
    bslS[p] = s;
  }
  if (tid >= 128 && tid < 192) {  // gate = local @ w_gate
    int p = tid - 128; float s = 0.f;
    for (int i = 0; i < LL; ++i) s += locS[i] * wGateG[i*PP + p];
    gateS[p] = s;
  }
  if (tid == 192) {               // backbone frame (e1,e2,e3,ca)
    const float* pp = posG + n*NA0*3;
    float nx=pp[0], ny=pp[1], nz=pp[2];
    float cax=pp[3], cay=pp[4], caz=pp[5];
    float cx=pp[6], cy=pp[7], cz=pp[8];
    float v1x=cx-cax, v1y=cy-cay, v1z=cz-caz;
    float r1 = rsqrtf(v1x*v1x+v1y*v1y+v1z*v1z + 1e-6f);
    float e1x=v1x*r1, e1y=v1y*r1, e1z=v1z*r1;
    float v2x=nx-cax, v2y=ny-cay, v2z=nz-caz;
    float d = e1x*v2x+e1y*v2y+e1z*v2z;
    float u2x=v2x-e1x*d, u2y=v2y-e1y*d, u2z=v2z-e1z*d;
    float r2 = rsqrtf(u2x*u2x+u2y*u2y+u2z*u2z + 1e-6f);
    float e2x=u2x*r2, e2y=u2y*r2, e2z=u2z*r2;
    float e3x=e1y*e2z-e1z*e2y, e3y=e1z*e2x-e1x*e2z, e3z=e1x*e2y-e1y*e2x;
    frm[0]=e1x; frm[1]=e1y; frm[2]=e1z;
    frm[3]=e2x; frm[4]=e2y; frm[5]=e2z;
    frm[6]=e3x; frm[7]=e3y; frm[8]=e3z;
    frm[9]=cax; frm[10]=cay; frm[11]=caz;
  }
  if (tid == 193) {               // mask count
    float c = 0.f;
    for (int s = 0; s < SS; ++s) c += mskS[s];
    cntS[0] = c;
  }
  __syncthreads();

  // ---- phase 3: pos_all = [pos ; rot @ p_local + ca] -------------------------
  if (tid < AA) {
    int a = tid;
    if (a < NA0) {
      for (int c = 0; c < 3; ++c) pallS[a*3+c] = posG[n*NA0*3 + a*3 + c];
    } else {
      int k = a - NA0;
      float px = plS[k*3+0], py = plS[k*3+1], pz = plS[k*3+2];
      for (int c = 0; c < 3; ++c)
        pallS[a*3+c] = frm[c]*px + frm[3+c]*py + frm[6+c]*pz + frm[9+c];
    }
  }
  __syncthreads();

  // ---- phase 4: build A (dirs|rbf) f16, B/W1/W2 f16 frag-swizzled ------------
  for (int i = tid; i < SS*AA; i += 256) {
    int s = i >> 4, a = i & 15;
    float rx = sposS[s*3+0] - pallS[a*3+0];
    float ry = sposS[s*3+1] - pallS[a*3+1];
    float rz = sposS[s*3+2] - pallS[a*3+2];
    float dist = sqrtf(rx*rx + ry*ry + rz*rz + 1e-6f);
    float inv = 1.0f / dist;
    _Float16* arow = aS + s*KA;
    arow[a*3+0] = (_Float16)(rx*inv);
    arow[a*3+1] = (_Float16)(ry*inv);
    arow[a*3+2] = (_Float16)(rz*inv);
    arow[48+a] = (_Float16)0.0f;                 // zero-pad cols 48..63
    v8h r0, r1;                                  // RBF: centers j*0.8, sigma 0.75
#pragma unroll
    for (int j = 0; j < 8; ++j) {
      float z0 = (dist - 0.8f*(float)j)     * (1.0f/0.75f);
      float z1 = (dist - 0.8f*(float)(j+8)) * (1.0f/0.75f);
      r0[j] = (_Float16)__expf(-z0*z0);
      r1[j] = (_Float16)__expf(-z1*z1);
    }
    *(v8h*)(arow + 64 + a*16)     = r0;          // 32B-aligned vector stores
    *(v8h*)(arow + 64 + a*16 + 8) = r1;
  }
  for (int i = tid; i < KA*PP; i += 256) {
    int k = i >> 6, p = i & 63;
    float v = (k < 48) ? wDirG[k*PP+p] : (k < 64) ? 0.0f : wDistG[(k-64)*PP+p];
    bS[bswz(k, p, PP)] = (_Float16)v;
  }
  for (int i = tid; i < PP*2*PP; i += 256) {     // w_mlp1 (64 x 128)
    int k = i >> 7, p = i & 127;
    w1S[bswz(k, p, 2*PP)] = (_Float16)wMlp1G[i];
  }
  for (int i = tid; i < 2*PP*PP; i += 256) {     // w_mlp2 (128 x 64)
    int k = i >> 6, p = i & 63;
    w2S[bswz(k, p, PP)] = (_Float16)wMlp2G[i];
  }
  __syncthreads();

  // ---- phase 5: GEMM1  pair = A(128x320) @ B(320x64) + type/local bias -------
  {
    const int m0 = wave * 16;
    const _Float16* aRow = aS + (m0 + lane15)*KA + koffA;
    const int rbase = m0 + ((lane < 16) ? 0 : 8);
    for (int nt = 0; nt < 4; ++nt) {
      const int p0 = nt * 16;
      const int colp = p0 + lane15;
      v8f acc = {};
#pragma unroll
      for (int kt = 0; kt < KA/32; ++kt) {
        v16h af = load_frag_a(aRow + kt*32);
        v16h bf = load_frag_b_swz(bS, kt, PP, colp, grp);
        acc = __builtin_amdgcn_wmma_f32_16x16x32_f16(false, af, false, bf,
                                                     (short)0, acc, false, false);
      }
#pragma unroll
      for (int r = 0; r < 8; ++r) {
        int row = rbase + r;
        pairF[row*PP + colp] = acc[r] + wtS[stS[row]*PP + colp] + bslS[colp];
      }
    }
  }
  __syncthreads();

  // ---- phase 6: LayerNorm over P=64 -> f16 -----------------------------------
  if (tid < SS) {
    const float* pr = pairF + tid*PP;
    float mu = 0.f;
    for (int p = 0; p < PP; ++p) mu += pr[p];
    mu *= (1.0f/PP);
    float var = 0.f;
    for (int p = 0; p < PP; ++p) { float d = pr[p]-mu; var += d*d; }
    var *= (1.0f/PP);
    float inv = rsqrtf(var + 1e-5f);
    for (int p = 0; p < PP; ++p)
      pH[tid*PP + p] = (_Float16)((pr[p]-mu)*inv*lnS[p] + lnO[p]);
  }
  __syncthreads();

  // ---- phase 7: MLP1  act = gelu(pairH(128x64) @ w1(64x128) + b1) ------------
  {
    const int m0 = wave * 16;
    const _Float16* aRow = pH + (m0 + lane15)*PP + koffA;
    const int rbase = m0 + ((lane < 16) ? 0 : 8);
    for (int nt = 0; nt < 8; ++nt) {
      const int p0 = nt * 16;
      const int colp = p0 + lane15;
      v8f acc = {};
#pragma unroll
      for (int kt = 0; kt < 2; ++kt) {
        v16h af = load_frag_a(aRow + kt*32);
        v16h bf = load_frag_b_swz(w1S, kt, 2*PP, colp, grp);
        acc = __builtin_amdgcn_wmma_f32_16x16x32_f16(false, af, false, bf,
                                                     (short)0, acc, false, false);
      }
#pragma unroll
      for (int r = 0; r < 8; ++r) {
        int row = rbase + r;
        actS[row*128 + colp] = (_Float16)gelu_f(acc[r] + b1S[colp]);
      }
    }
  }
  __syncthreads();

  // ---- phase 8: MLP2  pair = act(128x128) @ w2(128x64) + b2 ------------------
  {
    const int m0 = wave * 16;
    const _Float16* aRow = actS + (m0 + lane15)*128 + koffA;
    const int rbase = m0 + ((lane < 16) ? 0 : 8);
    for (int nt = 0; nt < 4; ++nt) {
      const int p0 = nt * 16;
      const int colp = p0 + lane15;
      v8f acc = {};
#pragma unroll
      for (int kt = 0; kt < 4; ++kt) {
        v16h af = load_frag_a(aRow + kt*32);
        v16h bf = load_frag_b_swz(w2S, kt, PP, colp, grp);
        acc = __builtin_amdgcn_wmma_f32_16x16x32_f16(false, af, false, bf,
                                                     (short)0, acc, false, false);
      }
#pragma unroll
      for (int r = 0; r < 8; ++r) {
        int row = rbase + r;
        pairF[row*PP + colp] = acc[r] + b2S[colp];
      }
    }
  }
  __syncthreads();

  // ---- phase 9: masked mean over S, gate -------------------------------------
  if (tid < PP) {
    float s = 0.f;
    for (int r = 0; r < SS; ++r) s += pairF[r*PP + tid] * mskS[r];
    float mean = s / fmaxf(cntS[0], 1.0f);
    hS[tid] = gelu_f(gateS[tid]) * mean;
  }
  __syncthreads();

  // ---- phase 10: out = h(64) @ w_out(64x256) ---------------------------------
  {
    float s = 0.f;
    for (int p = 0; p < PP; ++p) s += hS[p] * wOutG[p*LL + tid];
    outG[n*LL + tid] = s;
  }
}

extern "C" void kernel_launch(void* const* d_in, const int* in_sizes, int n_in,
                              void* d_out, int out_size, void* d_ws, size_t ws_size,
                              hipStream_t stream) {
  (void)in_sizes; (void)n_in; (void)d_ws; (void)ws_size; (void)out_size;
  smol_fused_kernel<<<dim3(NRES), dim3(256), SMEM_BYTES, stream>>>(
      (const float*)d_in[0],          // local
      (const float*)d_in[1],          // pos
      (const int*)d_in[2],            // smol_type
      (const float*)d_in[3],          // smol_pos
      (const unsigned char*)d_in[4],  // smol_mask (bool bytes)
      (const float*)d_in[5],          // w_points
      (const float*)d_in[6],          // w_type
      (const float*)d_in[7],          // w_local
      (const float*)d_in[8],          // w_dir
      (const float*)d_in[9],          // w_dist
      (const float*)d_in[10],         // ln_scale
      (const float*)d_in[11],         // ln_offset
      (const float*)d_in[12],         // w_mlp1
      (const float*)d_in[13],         // b_mlp1
      (const float*)d_in[14],         // w_mlp2
      (const float*)d_in[15],         // b_mlp2
      (const float*)d_in[16],         // w_gate
      (const float*)d_in[17],         // w_out
      (float*)d_out);
}
